// AttentionDenoiserLayer_86019605004396
// MI455X (gfx1250) — matrix operand order repaired and checked by reading, not compiled, not run on hardware
//
#include <hip/hip_runtime.h>
#include <hip/hip_bf16.h>

typedef __attribute__((ext_vector_type(16))) _Float16 v16h;
typedef __attribute__((ext_vector_type(8)))  float    v8f;
typedef unsigned int u32x4 __attribute__((ext_vector_type(4)));
typedef int          i32x8 __attribute__((ext_vector_type(8)));
typedef int          i32x4 __attribute__((ext_vector_type(4)));

#define H_DIM    128
#define N_HEADS  8
#define HEAD_DIM 16
#define N_ATOMS  2048
#define N_RES    256
#define C_Z      128
#define FF_DIM   512

#if defined(__has_builtin)
#  if __has_builtin(__builtin_amdgcn_tensor_load_to_lds)
#    define HAVE_TDM 1
#  endif
#endif
#ifndef HAVE_TDM
#  define HAVE_TDM 0
#endif

// ---------------------------------------------------------------------------
// TDM: DMA a 16-row x 32-col f16 tile from a row-major [rows?,K] matrix into
// LDS.  D# packing per CDNA5 ISA 8.3/8.4: group0 {count=1, lds_addr,
// global_addr[56:0], type=2}; group1 {data_size=1(2B), tensor_dim0=K,
// tensor_dim1=16, tile_dim0=32, tile_dim1=16, tensor_dim0_stride=K}.
// ---------------------------------------------------------------------------
#if HAVE_TDM
__device__ __forceinline__ void tdm_load_tile16x32(unsigned lds_off,
                                                   const _Float16* gptr, int K) {
  unsigned long long ga = (unsigned long long)(uintptr_t)gptr;
  u32x4 g0;
  g0[0] = 1u;                                           // count=1, user mode
  g0[1] = lds_off;                                      // lds_addr (bytes)
  g0[2] = (unsigned)(ga & 0xffffffffu);                 // global_addr[31:0]
  g0[3] = (unsigned)((ga >> 32) & 0x1ffffffu) | (2u << 30);  // ga[56:32]|type=2
  unsigned td0 = (unsigned)K;                           // tensor_dim0 (elems)
  unsigned td1 = 16u;                                   // tensor_dim1 (rows)
  unsigned long long st0 = (unsigned long long)K;       // dim0 stride (elems)
  i32x8 g1;
  g1[0] = (int)(1u << 16);                              // data_size=1 -> 2B
  g1[1] = (int)((td0 & 0xffffu) << 16);                 // abar=0 | td0.lo
  g1[2] = (int)((td0 >> 16) | ((td1 & 0xffffu) << 16)); // td0.hi | td1.lo
  g1[3] = (int)((td1 >> 16) | (32u << 16));             // td1.hi | tile_dim0=32
  g1[4] = (int)16u;                                     // tile_dim1=16, tile_dim2=0
  g1[5] = (int)(unsigned)(st0 & 0xffffffffu);           // stride0[31:0]
  g1[6] = (int)(unsigned)((st0 >> 32) & 0xffffu);       // stride0[47:32]
  g1[7] = 0;
  i32x4 gz = {0, 0, 0, 0};
#if __clang_major__ >= 23
  i32x8 gz8 = {0, 0, 0, 0, 0, 0, 0, 0};
  __builtin_amdgcn_tensor_load_to_lds(g0, g1, gz, gz, gz8, 0);
#else
  __builtin_amdgcn_tensor_load_to_lds(g0, g1, gz, gz, 0);
#endif
}
__device__ __forceinline__ unsigned lds_offset_of(const void* p) {
  return (unsigned)(unsigned long long)(uintptr_t)p;  // LDS aperture: addr[31:0]
}
#endif

// ---------------------------------------------------------------------------
// f32 -> f16 convert with scale, source-column padding (c >= src_cols -> 0)
// ---------------------------------------------------------------------------
__global__ void k_cvt(const float* __restrict__ src, _Float16* __restrict__ dst,
                      int rows, int cols, int src_cols, int src_ld, int dst_ld,
                      float scale) {
  int idx = blockIdx.x * blockDim.x + threadIdx.x;
  if (idx >= rows * cols) return;
  int r = idx / cols, c = idx % cols;
  float v = (c < src_cols) ? src[r * src_ld + c] * scale : 0.0f;
  dst[r * dst_ld + c] = (_Float16)v;
}

// ---------------------------------------------------------------------------
// f32 weights [K,N] -> f16 swizzled into the contiguous-k B-fragment order used here:
// dst[((nt*(K/32) + kt)*32 + lane)*16 + e] = W[kt*32 + e + (lane>>4)*16, nt*16 + lane%16]
// so each lane reads its 16 B elements as one contiguous 32B chunk.
// ---------------------------------------------------------------------------
__global__ void k_cvt_sw(const float* __restrict__ src, _Float16* __restrict__ dst,
                         int K, int N, int src_cols, int src_ld, float scale) {
  int idx = blockIdx.x * blockDim.x + threadIdx.x;
  if (idx >= K * N) return;
  int e    = idx & 15;
  int lane = (idx >> 4) & 31;
  int kt   = (idx >> 9) % (K >> 5);
  int nt   = (idx >> 9) / (K >> 5);
  int colg = nt * 16 + (lane & 15);
  int krow = kt * 32 + e + ((lane >> 4) << 4);
  float v = (colg < src_cols) ? src[krow * src_ld + colg] * scale : 0.0f;
  dst[idx] = (_Float16)v;
}

// f32 copy with scale + zero padding past src_n
__global__ void k_copyf(const float* __restrict__ s, float* __restrict__ d,
                        int n, int src_n, float scale) {
  int i = blockIdx.x * blockDim.x + threadIdx.x;
  if (i < n) d[i] = (i < src_n) ? s[i] * scale : 0.0f;
}

// ---------------------------------------------------------------------------
// LayerNorm over 128 cols, one row per 128-thread block, f16 output
// ---------------------------------------------------------------------------
__global__ void k_ln(const float* __restrict__ x, const float* __restrict__ g,
                     const float* __restrict__ b, _Float16* __restrict__ out) {
  __shared__ float red[128];
  int row = blockIdx.x, tid = threadIdx.x;
  float v = x[row * 128 + tid];
  red[tid] = v; __syncthreads();
  for (int s = 64; s > 0; s >>= 1) { if (tid < s) red[tid] += red[tid + s]; __syncthreads(); }
  float mean = red[0] * (1.0f / 128.0f);
  __syncthreads();
  float d = v - mean;
  red[tid] = d * d; __syncthreads();
  for (int s = 64; s > 0; s >>= 1) { if (tid < s) red[tid] += red[tid + s]; __syncthreads(); }
  float inv = rsqrtf(red[0] * (1.0f / 128.0f) + 1e-5f);
  out[row * 128 + tid] = (_Float16)(d * inv * g[tid] + b[tid]);
}

// ---------------------------------------------------------------------------
// Generic WMMA GEMM: out[M,N] = A[M,K] @ Bsw[K,N] + bias (+GELU) (+resid)
// A tiles (16x32 f16) staged to LDS by the Tensor Data Mover, double-buffered;
// B is pre-swizzled into fragment order (contiguous 32B per lane per k-step).
// One 16-row M tile per workgroup; one 16x16 C tile per wave.
// ---------------------------------------------------------------------------
template <bool GELU, bool RES, bool OUT16>
__global__ void k_gemm(const _Float16* __restrict__ A, const _Float16* __restrict__ Bsw,
                       const float* __restrict__ bias, const float* __restrict__ resid,
                       void* __restrict__ out, int M, int N, int K) {
  __shared__ _Float16 As[2][16 * 32];
  const int lane   = threadIdx.x & 31;
  const int wave   = threadIdx.x >> 5;
  const int nwaves = blockDim.x >> 5;
  const int m0  = blockIdx.x * 16;
  int n0 = (blockIdx.y * nwaves + wave) * 16;
  if (n0 + 16 > N) n0 = N - 16;           // never triggered for our launches
  const int col = lane & 15;
  const int hi  = lane >> 4;
  const int ktiles = K >> 5;

#if HAVE_TDM
  if (wave == 0) {
    tdm_load_tile16x32(lds_offset_of(&As[0][0]), A + (size_t)m0 * K, K);
    __builtin_amdgcn_s_wait_tensorcnt(0);
  }
  __syncthreads();
#endif

  v8f c = {};
  for (int kt = 0; kt < ktiles; ++kt) {
    const int cur = kt & 1;
#if HAVE_TDM
    if (wave == 0 && kt + 1 < ktiles)     // prefetch next tile into other buffer
      tdm_load_tile16x32(lds_offset_of(&As[cur ^ 1][0]),
                         A + (size_t)m0 * K + (kt + 1) * 32, K);
#else
    __syncthreads();
    for (int t = threadIdx.x; t < 256; t += blockDim.x) {
      int m = t >> 4, c2 = t & 15;
      ((unsigned int*)&As[cur][0])[t] =
          *(const unsigned int*)(A + (size_t)(m0 + m) * K + kt * 32 + c2 * 2);
    }
    __syncthreads();
#endif
    // A fragment: row = lane%16, K = e + (lane/16)*16
    union { v16h v; unsigned int u[8]; } a;
    #pragma unroll
    for (int vv = 0; vv < 8; ++vv) {
      int e  = vv * 2;
      int kl = e + hi * 16;
      a.u[vv] = *(const unsigned int*)(&As[cur][col * 32 + kl]);
    }
    // B fragment: contiguous 32B per lane from swizzled weights
    union { v16h v; unsigned int u[8]; } bf;
    const unsigned int* bp32 = (const unsigned int*)(Bsw +
        ((((size_t)(n0 >> 4)) * ktiles + kt) * 32 + lane) * 16);
    #pragma unroll
    for (int vv = 0; vv < 8; ++vv) bf.u[vv] = bp32[vv];
    c = __builtin_amdgcn_wmma_f32_16x16x32_f16(false, a.v, false, bf.v,
                                               (short)0, c, false, false);
#if HAVE_TDM
    __syncthreads();                      // everyone done reading As[cur]
    if (wave == 0 && kt + 1 < ktiles) __builtin_amdgcn_s_wait_tensorcnt(0);
    __syncthreads();                      // publish As[cur^1]
#endif
  }
  float bcol = bias[n0 + col];
  #pragma unroll
  for (int r = 0; r < 8; ++r) {
    int row = m0 + r + hi * 8;
    float val = c[r] + bcol;
    if (GELU) val = 0.5f * val * (1.0f + erff(val * 0.70710678118654752f));
    if (RES)  val += resid[(size_t)row * N + n0 + col];
    if (OUT16) ((_Float16*)out)[(size_t)row * N + n0 + col] = (_Float16)val;
    else       ((float*)out)[(size_t)row * N + n0 + col] = val;
  }
}

// ---------------------------------------------------------------------------
// Streaming attention, one wave per (16-query tile, head).
// qkv: [2048, 384] f16 (q|k|v per row, softmax scale folded into q).
// pb:  [N_RES*N_RES, 16] f16 pair bias (head in low 8 cols).
// ---------------------------------------------------------------------------
__global__ void k_attn(const _Float16* __restrict__ qkv,
                       const _Float16* __restrict__ pb,
                       const int* __restrict__ a2r,
                       const unsigned char* __restrict__ mask,
                       _Float16* __restrict__ outp) {
  __shared__ _Float16 lp[16 * 32];
  int lane  = threadIdx.x & 31;
  int col   = lane & 15;
  int hi    = lane >> 4;
  int itile = blockIdx.x;
  int head  = blockIdx.y;

  // q fragment: row = lane%16, real K(d) = e for lanes 0..15, lanes 16..31 zero-padded
  union { v16h v; unsigned int u[8]; } qa;
  #pragma unroll
  for (int vv = 0; vv < 8; ++vv) qa.u[vv] = 0u;
  if (hi == 0) {
    const _Float16* qrow = qkv + (size_t)(itile * 16 + col) * 384 + head * 16;
    #pragma unroll
    for (int vv = 0; vv < 8; ++vv) qa.u[vv] = *(const unsigned int*)(qrow + vv * 2);
  }
  int res_i[8];
  #pragma unroll
  for (int r = 0; r < 8; ++r) res_i[r] = a2r[itile * 16 + r + hi * 8];

  v8f o = {};
  float l[8], mrow[8];
  #pragma unroll
  for (int r = 0; r < 8; ++r) { l[r] = 0.0f; mrow[r] = -1e30f; }
  v8f zc = {};

  for (int jb = 0; jb < N_ATOMS; jb += 32) {
    float s0[8], s1[8];
    #pragma unroll
    for (int t = 0; t < 2; ++t) {
      int j0 = jb + t * 16;
      // k^T as B: col = key (lane%16); lanes 16..31 cover zero-padded K
      union { v16h v; unsigned int u[8]; } kb;
      #pragma unroll
      for (int vv = 0; vv < 8; ++vv) kb.u[vv] = 0u;
      if (hi == 0) {
        const _Float16* krow = qkv + (size_t)(j0 + col) * 384 + 128 + head * 16;
        #pragma unroll
        for (int vv = 0; vv < 8; ++vv) kb.u[vv] = *(const unsigned int*)(krow + vv * 2);
      }
      v8f s = __builtin_amdgcn_wmma_f32_16x16x32_f16(false, qa.v, false, kb.v,
                                                     (short)0, zc, false, false);
      int j = j0 + col;
      int rj = a2r[j];
      // branchless masking: unconditional bias load + additive penalty
      float neg = (mask[j] != 0) ? 0.0f : -3.0e30f;
      const _Float16* pbrow = pb + (size_t)rj * 16 + head;
      #pragma unroll
      for (int r = 0; r < 8; ++r) {
        float bz = (float)pbrow[(size_t)res_i[r] * (N_RES * 16)];
        float sv = s[r] + bz + neg;
        if (t == 0) s0[r] = sv; else s1[r] = sv;
      }
    }
    // online softmax bookkeeping (rows live in C-layout: row = r + hi*8)
    float p0[8], p1[8];
    #pragma unroll
    for (int r = 0; r < 8; ++r) {
      float mx = fmaxf(s0[r], s1[r]);
      #pragma unroll
      for (int m = 8; m >= 1; m >>= 1) mx = fmaxf(mx, __shfl_xor(mx, m, 16));
      float mn = fmaxf(mrow[r], mx);
      float alpha = __expf(mrow[r] - mn);
      mrow[r] = mn;
      p0[r] = __expf(s0[r] - mn);
      p1[r] = __expf(s1[r] - mn);
      float su = p0[r] + p1[r];
      #pragma unroll
      for (int m = 8; m >= 1; m >>= 1) su += __shfl_xor(su, m, 16);
      l[r] = l[r] * alpha + su;
      o[r] = o[r] * alpha;
    }
    // P tile (16x32) through LDS to pick up the A-matrix layout
    #pragma unroll
    for (int r = 0; r < 8; ++r) {
      int m = r + hi * 8;
      lp[m * 32 + col]      = (_Float16)p0[r];
      lp[m * 32 + 16 + col] = (_Float16)p1[r];
    }
    __syncthreads();
    union { v16h v; unsigned int u[8]; } pa;
    #pragma unroll
    for (int vv = 0; vv < 8; ++vv) {
      int e  = vv * 2;
      int kl = e + hi * 16;
      pa.u[vv] = *(const unsigned int*)(&lp[col * 32 + kl]);
    }
    // V as B: col = head dim d (lane%16), K = key offset e + hi*16
    union { v16h v; _Float16 h[16]; } vb;
    #pragma unroll
    for (int e = 0; e < 16; ++e) {
      int j = jb + e + hi * 16;
      vb.h[e] = qkv[(size_t)j * 384 + 256 + head * 16 + col];
    }
    o = __builtin_amdgcn_wmma_f32_16x16x32_f16(false, pa.v, false, vb.v,
                                               (short)0, o, false, false);
    __syncthreads();
  }
  #pragma unroll
  for (int r = 0; r < 8; ++r) {
    int i = itile * 16 + r + hi * 8;
    outp[(size_t)i * H_DIM + head * 16 + col] = (_Float16)(o[r] / l[r]);
  }
}

// ---------------------------------------------------------------------------
// Final coordinate update: x_new = x + LN(h) @ Wc + bc, one row per block
// ---------------------------------------------------------------------------
__global__ void k_coords(const float* __restrict__ h, const float* __restrict__ x,
                         const float* __restrict__ g, const float* __restrict__ b,
                         const float* __restrict__ Wc, const float* __restrict__ bc,
                         float* __restrict__ xout) {
  __shared__ float red[128];
  __shared__ float sm[2];
  int row = blockIdx.x, tid = threadIdx.x;
  float v = h[row * 128 + tid];
  red[tid] = v; __syncthreads();
  for (int s = 64; s > 0; s >>= 1) { if (tid < s) red[tid] += red[tid + s]; __syncthreads(); }
  if (tid == 0) sm[0] = red[0] * (1.0f / 128.0f);
  __syncthreads();
  float d = v - sm[0];
  red[tid] = d * d; __syncthreads();
  for (int s = 64; s > 0; s >>= 1) { if (tid < s) red[tid] += red[tid + s]; __syncthreads(); }
  if (tid == 0) sm[1] = rsqrtf(red[0] * (1.0f / 128.0f) + 1e-5f);
  __syncthreads();
  float hn = d * sm[1] * g[tid] + b[tid];
  for (int j = 0; j < 3; ++j) {
    __syncthreads();
    red[tid] = hn * Wc[tid * 3 + j];
    __syncthreads();
    for (int s = 64; s > 0; s >>= 1) { if (tid < s) red[tid] += red[tid + s]; __syncthreads(); }
    if (tid == 0) xout[row * 3 + j] = x[row * 3 + j] + red[0] + bc[j];
  }
}

// ---------------------------------------------------------------------------
extern "C" void kernel_launch(void* const* d_in, const int* in_sizes, int n_in,
                              void* d_out, int out_size, void* d_ws, size_t ws_size,
                              hipStream_t stream) {
  (void)in_sizes; (void)n_in; (void)out_size; (void)ws_size;
  const float* h     = (const float*)d_in[0];
  const float* x     = (const float*)d_in[1];
  const float* z     = (const float*)d_in[2];
  const int*   a2r   = (const int*)d_in[3];
  const unsigned char* mask = (const unsigned char*)d_in[4];
  const float* ln1_g = (const float*)d_in[5];
  const float* ln1_b = (const float*)d_in[6];
  const float* Wq = (const float*)d_in[7];  const float* bq = (const float*)d_in[8];
  const float* Wk = (const float*)d_in[9];  const float* bk = (const float*)d_in[10];
  const float* Wv = (const float*)d_in[11]; const float* bv = (const float*)d_in[12];
  const float* Wo = (const float*)d_in[13]; const float* bo = (const float*)d_in[14];
  const float* Wp = (const float*)d_in[15]; const float* bp = (const float*)d_in[16];
  const float* ln2_g = (const float*)d_in[17];
  const float* ln2_b = (const float*)d_in[18];
  const float* W1 = (const float*)d_in[19]; const float* b1 = (const float*)d_in[20];
  const float* W2 = (const float*)d_in[21]; const float* b2 = (const float*)d_in[22];
  const float* cln_g = (const float*)d_in[23];
  const float* cln_b = (const float*)d_in[24];
  const float* Wc = (const float*)d_in[25]; const float* bc = (const float*)d_in[26];

  char* ws = (char*)d_ws;
  size_t off = 0;
  auto alloc = [&](size_t bytes) -> void* {
    void* p = ws + off;
    off = (off + bytes + 255) & ~(size_t)255;
    return p;
  };
  _Float16* hn16   = (_Float16*)alloc((size_t)N_ATOMS * H_DIM * 2);
  _Float16* qkv16  = (_Float16*)alloc((size_t)N_ATOMS * 384 * 2);
  _Float16* ao16   = (_Float16*)alloc((size_t)N_ATOMS * H_DIM * 2);
  float*    h1     = (float*)   alloc((size_t)N_ATOMS * H_DIM * 4);
  _Float16* ln216  = (_Float16*)alloc((size_t)N_ATOMS * H_DIM * 2);
  _Float16* act16  = (_Float16*)alloc((size_t)N_ATOMS * FF_DIM * 2);
  _Float16* z16    = (_Float16*)alloc((size_t)N_RES * N_RES * C_Z * 2);
  _Float16* pb16   = (_Float16*)alloc((size_t)N_RES * N_RES * 16 * 2);
  _Float16* wqkv16 = (_Float16*)alloc((size_t)128 * 384 * 2);  // swizzled
  _Float16* wo16   = (_Float16*)alloc((size_t)128 * 128 * 2);  // swizzled
  _Float16* w116   = (_Float16*)alloc((size_t)128 * 512 * 2);  // swizzled
  _Float16* w216   = (_Float16*)alloc((size_t)512 * 128 * 2);  // swizzled
  _Float16* wp16   = (_Float16*)alloc((size_t)128 * 16 * 2);   // swizzled, padded
  float*    bqkv   = (float*)   alloc(384 * 4);
  float*    bppad  = (float*)   alloc(16 * 4);

  const float qscale = 0.25f;  // 1/sqrt(HEAD_DIM), folded into Wq/bq
  int T;
  // weight conversions, swizzled to B-fragment order (f32 -> f16)
  T = 128 * 128;  // each 128x128 weight occupies 16384 swizzled halves
  k_cvt_sw<<<(T + 255) / 256, 256, 0, stream>>>(Wq, wqkv16,             128, 128, 128, 128, qscale);
  k_cvt_sw<<<(T + 255) / 256, 256, 0, stream>>>(Wk, wqkv16 + 128 * 128, 128, 128, 128, 128, 1.0f);
  k_cvt_sw<<<(T + 255) / 256, 256, 0, stream>>>(Wv, wqkv16 + 256 * 128, 128, 128, 128, 128, 1.0f);
  k_cvt_sw<<<(T + 255) / 256, 256, 0, stream>>>(Wo, wo16,               128, 128, 128, 128, 1.0f);
  T = 128 * 512;
  k_cvt_sw<<<(T + 255) / 256, 256, 0, stream>>>(W1, w116, 128, 512, 512, 512, 1.0f);
  k_cvt_sw<<<(T + 255) / 256, 256, 0, stream>>>(W2, w216, 512, 128, 128, 128, 1.0f);
  T = 128 * 16;
  k_cvt_sw<<<(T + 255) / 256, 256, 0, stream>>>(Wp, wp16, 128, 16, 8, 8, 1.0f);  // pad N 8->16
  k_copyf<<<1, 128, 0, stream>>>(bq, bqkv,       128, 128, qscale);
  k_copyf<<<1, 128, 0, stream>>>(bk, bqkv + 128, 128, 128, 1.0f);
  k_copyf<<<1, 128, 0, stream>>>(bv, bqkv + 256, 128, 128, 1.0f);
  k_copyf<<<1, 32,  0, stream>>>(bp, bppad, 16, 8, 1.0f);

  // z -> f16 (plain row-major; consumed as the GEMM A operand via TDM)
  T = N_RES * N_RES * C_Z;
  k_cvt<<<(T + 255) / 256, 256, 0, stream>>>(z, z16, N_RES * N_RES, C_Z, C_Z, C_Z, C_Z, 1.0f);

  // LN1
  k_ln<<<N_ATOMS, 128, 0, stream>>>(h, ln1_g, ln1_b, hn16);

  // fused QKV projection: [2048,128] @ [128,384]
  k_gemm<false, false, true><<<dim3(N_ATOMS / 16, 3), 256, 0, stream>>>(
      hn16, wqkv16, bqkv, nullptr, qkv16, N_ATOMS, 384, 128);

  // pair bias: [65536,128] @ [128,16]
  k_gemm<false, false, true><<<dim3(N_RES * N_RES / 16, 1), 32, 0, stream>>>(
      z16, wp16, bppad, nullptr, pb16, N_RES * N_RES, 16, C_Z);

  // streaming attention
  k_attn<<<dim3(N_ATOMS / 16, N_HEADS), 32, 0, stream>>>(qkv16, pb16, a2r, mask, ao16);

  // output projection + residual: h1 = h + ao @ Wo + bo
  k_gemm<false, true, false><<<dim3(N_ATOMS / 16, 1), 256, 0, stream>>>(
      ao16, wo16, bo, h, h1, N_ATOMS, H_DIM, H_DIM);

  // LN2
  k_ln<<<N_ATOMS, 128, 0, stream>>>(h1, ln2_g, ln2_b, ln216);

  // FFN1 + exact GELU
  k_gemm<true, false, true><<<dim3(N_ATOMS / 16, 4), 256, 0, stream>>>(
      ln216, w116, b1, nullptr, act16, N_ATOMS, FF_DIM, H_DIM);

  // FFN2 + residual -> final h (first part of d_out)
  float* h_out = (float*)d_out;
  k_gemm<false, true, false><<<dim3(N_ATOMS / 16, 1), 256, 0, stream>>>(
      act16, w216, b2, h1, h_out, N_ATOMS, H_DIM, FF_DIM);

  // coordinate update -> second part of d_out
  float* x_out = h_out + (size_t)N_ATOMS * H_DIM;
  k_coords<<<N_ATOMS, 128, 0, stream>>>(h_out, x, cln_g, cln_b, Wc, bc, x_out);
}
